// SingleHeadAttention_6107443495049
// MI455X (gfx1250) — compile-verified, matrix operand order checked by reading, not run on hardware
//
#include <hip/hip_runtime.h>
#include <hip/hip_bf16.h>

// ---- CDNA5 WMMA vector types ------------------------------------------------
typedef _Float16 v16h __attribute__((ext_vector_type(16)));
typedef _Float16 h8   __attribute__((ext_vector_type(8)));
typedef float    v8f  __attribute__((ext_vector_type(8)));

// Problem constants (match the reference)
constexpr int Bc = 4, Tc = 4096, Cc = 1024, Hc = 256;
constexpr float SCALE = 0.0625f;          // 1/sqrt(H) = 1/16
constexpr float NEG_BIG = -1.0e30f;       // "-inf" that stays NaN-free in exp

__device__ __forceinline__ v8f wmma_f16(const v16h& a, const v16h& b, const v8f& c) {
    return __builtin_amdgcn_wmma_f32_16x16x32_f16(false, a, false, b, (short)0, c,
                                                  false, false);
}

// =============================================================================
// Kernel 0: convert W (C x H, f32) -> Wt (H x C, f16).  Bandwidth-trivial.
// Thread mapping: h fastest -> coalesced reads of W; writes strided (fine).
// =============================================================================
__global__ __launch_bounds__(256)
void wconv_kernel(const float* __restrict__ W, _Float16* __restrict__ Wt) {
    const int idx = blockIdx.x * blockDim.x + threadIdx.x;   // 0 .. C*H-1
    const int c = idx / Hc;
    const int h = idx % Hc;
    Wt[(size_t)h * Cc + c] = (_Float16)W[idx];
}

// =============================================================================
// Kernel 1: projection GEMM  E[BT,C] x W[C,H] -> 16 rows x 256 cols per wave.
// One wave owns a 16-row strip and ALL 16 column tiles (16 accumulators), so
// each A-fragment load feeds 16 WMMAs. B-fragments come from the f16
// transposed Wt: two contiguous 16-byte loads per tile per K-chunk.
// MAT: 0 -> Qh[BT][H], 1 -> Kh[BT][H], 2 -> Vt[B][H][T] (transposed store).
// =============================================================================
template <int MAT>
__global__ __launch_bounds__(256)
void proj_strip_kernel(const float* __restrict__ E,
                       const _Float16* __restrict__ Wt,   // [H][C] f16
                       _Float16* __restrict__ Out) {      // Qh/Kh or Vt
    const int lane  = threadIdx.x & 31;
    const int gwave = (blockIdx.x * blockDim.x + threadIdx.x) >> 5;
    const int nTilesM = (Bc * Tc) / 16;                   // 1024
    if (gwave >= nTilesM) return;                         // wave-uniform

    const int m0  = gwave * 16;
    const int l15 = lane & 15;
    const int hlf = lane >> 4;
    const int khi = hlf * 8;

    const float* arow = E + (size_t)(m0 + l15) * Cc;

    v8f acc[16];
    #pragma unroll
    for (int nt = 0; nt < 16; ++nt) acc[nt] = (v8f){};

    for (int kc = 0; kc < Cc / 32; ++kc) {
        const int k0 = kc * 32;
        // ---- A fragment (16x32 f16): contiguous float4 loads, convert ------
        float4 f0 = *(const float4*)(arow + k0 + khi);
        float4 f1 = *(const float4*)(arow + k0 + khi + 4);
        float4 f2 = *(const float4*)(arow + k0 + 16 + khi);
        float4 f3 = *(const float4*)(arow + k0 + 16 + khi + 4);
        v16h a;
        a[0]=(_Float16)f0.x; a[1]=(_Float16)f0.y; a[2]=(_Float16)f0.z; a[3]=(_Float16)f0.w;
        a[4]=(_Float16)f1.x; a[5]=(_Float16)f1.y; a[6]=(_Float16)f1.z; a[7]=(_Float16)f1.w;
        a[8]=(_Float16)f2.x; a[9]=(_Float16)f2.y; a[10]=(_Float16)f2.z; a[11]=(_Float16)f2.w;
        a[12]=(_Float16)f3.x; a[13]=(_Float16)f3.y; a[14]=(_Float16)f3.z; a[15]=(_Float16)f3.w;

        // ---- 16 column tiles share this A fragment --------------------------
        #pragma unroll
        for (int nt = 0; nt < 16; ++nt) {
            const _Float16* wrow = Wt + (size_t)(nt * 16 + l15) * Cc + k0;
            h8 lo = *(const h8*)(wrow + khi);
            h8 hi = *(const h8*)(wrow + 16 + khi);
            v16h bfrag;
            #pragma unroll
            for (int e = 0; e < 8; ++e) { bfrag[e] = lo[e]; bfrag[8 + e] = hi[e]; }
            acc[nt] = wmma_f16(a, bfrag, acc[nt]);
        }
    }

    // ---- store (lane L, vgpr v -> M = v + 8*hlf, N = l15) -------------------
    #pragma unroll
    for (int nt = 0; nt < 16; ++nt) {
        const int col = nt * 16 + l15;
        #pragma unroll
        for (int v = 0; v < 8; ++v) {
            const int row = m0 + v + 8 * hlf;
            const _Float16 hv = (_Float16)acc[nt][v];
            if (MAT < 2) {
                Out[(size_t)row * Hc + col] = hv;             // Qh / Kh row-major
            } else {
                const int bb = row / Tc, tt = row % Tc;
                Out[((size_t)bb * Hc + col) * Tc + tt] = hv;  // Vt transposed
            }
        }
    }
}

// =============================================================================
// Kernel 2: causal flash attention. One wave per 16-query tile.
//   S(16x32) = Q(16x256) x K_blk^T      -> 16 wmma / key block
//   online softmax (f32, cross-lane reductions over the 16-lane N groups)
//   P -> LDS transpose (C/D layout -> A layout), s_wait_dscnt
//   O += P(16x32) x V_blk(32x256)       -> 16 wmma / key block
// =============================================================================
__global__ __launch_bounds__(256)
void flash_attn_kernel(const _Float16* __restrict__ Qh,
                       const _Float16* __restrict__ Kh,
                       const _Float16* __restrict__ Vt,
                       float* __restrict__ out) {
    __shared__ __align__(16) _Float16 plds[8][16 * 32];   // per-wave P staging

    const int lane = threadIdx.x & 31;
    const int wv   = threadIdx.x >> 5;
    const int gw   = blockIdx.x * 8 + wv;
    const int tiles = (Bc * Tc) / 16;                      // 1024
    if (gw >= tiles) return;                               // wave-uniform

    const int b  = gw / (Tc / 16);
    const int q0 = (gw % (Tc / 16)) * 16;
    const int l15 = lane & 15;
    const int hlf = lane >> 4;
    const int khi = hlf * 8;

    // ---- Q fragments for all 8 H-chunks, kept in registers ------------------
    v16h qf[8];
    {
        const _Float16* qrow = Qh + ((size_t)b * Tc + q0 + l15) * Hc;
        #pragma unroll
        for (int hc = 0; hc < 8; ++hc) {
            h8 lo = *(const h8*)(qrow + hc * 32 + khi);
            h8 hi = *(const h8*)(qrow + hc * 32 + 16 + khi);
            #pragma unroll
            for (int e = 0; e < 8; ++e) { qf[hc][e] = lo[e]; qf[hc][8 + e] = hi[e]; }
        }
    }

    v8f o[16];
    #pragma unroll
    for (int t = 0; t < 16; ++t) o[t] = (v8f){};
    v8f mrow, lrow;
    #pragma unroll
    for (int e = 0; e < 8; ++e) { mrow[e] = NEG_BIG; lrow[e] = 0.0f; }

    const int qhi = q0 + 15;
    for (int j = 0; j <= qhi; j += 32) {                   // causal key blocks
        // prefetch next K block into cache hierarchy
        __builtin_prefetch(Kh + ((size_t)b * Tc + j + 32) * Hc, 0, 1);

        // ---- S = Q x K^T for the two 16-column halves of this block --------
        v8f s0 = {}, s1 = {};
        const _Float16* k0row = Kh + ((size_t)b * Tc + j + l15) * Hc;
        const _Float16* k1row = k0row + (size_t)16 * Hc;
        #pragma unroll
        for (int hc = 0; hc < 8; ++hc) {
            v16h kb0, kb1;
            {
                h8 lo = *(const h8*)(k0row + hc * 32 + khi);
                h8 hi = *(const h8*)(k0row + hc * 32 + 16 + khi);
                #pragma unroll
                for (int e = 0; e < 8; ++e) { kb0[e] = lo[e]; kb0[8 + e] = hi[e]; }
            }
            {
                h8 lo = *(const h8*)(k1row + hc * 32 + khi);
                h8 hi = *(const h8*)(k1row + hc * 32 + 16 + khi);
                #pragma unroll
                for (int e = 0; e < 8; ++e) { kb1[e] = lo[e]; kb1[8 + e] = hi[e]; }
            }
            s0 = wmma_f16(qf[hc], kb0, s0);
            s1 = wmma_f16(qf[hc], kb1, s1);
        }

        // ---- scale + causal mask (element e: row M = e + 8*hlf) ------------
        #pragma unroll
        for (int e = 0; e < 8; ++e) {
            const int qr  = q0 + e + 8 * hlf;
            const int kc0 = j + l15;
            s0[e] = (kc0      <= qr) ? s0[e] * SCALE : NEG_BIG;
            s1[e] = (kc0 + 16 <= qr) ? s1[e] * SCALE : NEG_BIG;
        }

        // ---- online softmax: row max over 32 cols ---------------------------
        v8f rm;
        #pragma unroll
        for (int e = 0; e < 8; ++e) rm[e] = fmaxf(s0[e], s1[e]);
        #pragma unroll
        for (int off = 1; off < 16; off <<= 1)
            #pragma unroll
            for (int e = 0; e < 8; ++e)
                rm[e] = fmaxf(rm[e], __shfl_xor(rm[e], off, 32));

        v8f mnew, p0, p1, corr;
        #pragma unroll
        for (int e = 0; e < 8; ++e) {
            mnew[e] = fmaxf(mrow[e], rm[e]);
            p0[e]   = __expf(s0[e] - mnew[e]);
            p1[e]   = __expf(s1[e] - mnew[e]);
            corr[e] = __expf(mrow[e] - mnew[e]);
        }
        v8f rs;
        #pragma unroll
        for (int e = 0; e < 8; ++e) rs[e] = p0[e] + p1[e];
        #pragma unroll
        for (int off = 1; off < 16; off <<= 1)
            #pragma unroll
            for (int e = 0; e < 8; ++e)
                rs[e] += __shfl_xor(rs[e], off, 32);
        #pragma unroll
        for (int e = 0; e < 8; ++e) {
            lrow[e] = lrow[e] * corr[e] + rs[e];
            mrow[e] = mnew[e];
        }
        #pragma unroll
        for (int t = 0; t < 16; ++t)
            #pragma unroll
            for (int e = 0; e < 8; ++e) o[t][e] *= corr[e];

        // ---- P: C/D layout -> LDS -> A-fragment layout ----------------------
        #pragma unroll
        for (int e = 0; e < 8; ++e) {
            const int m = e + 8 * hlf;
            plds[wv][m * 32 + l15]      = (_Float16)p0[e];
            plds[wv][m * 32 + 16 + l15] = (_Float16)p1[e];
        }
        asm volatile("s_wait_dscnt 0" ::: "memory");   // wave-private: DS in-order
        v16h pa;
        #pragma unroll
        for (int e = 0; e < 8; ++e) {
            pa[e]     = plds[wv][l15 * 32 + khi + e];
            pa[8 + e] = plds[wv][l15 * 32 + 16 + khi + e];
        }
        asm volatile("s_wait_dscnt 0" ::: "memory");   // loads done before re-store

        // ---- O += P x V for each 16-wide H tile -----------------------------
        #pragma unroll
        for (int ht = 0; ht < 16; ++ht) {
            const _Float16* vrow = Vt + ((size_t)b * Hc + ht * 16 + l15) * Tc + j;
            h8 lo = *(const h8*)(vrow + khi);
            h8 hi = *(const h8*)(vrow + 16 + khi);
            v16h vb;
            #pragma unroll
            for (int e = 0; e < 8; ++e) { vb[e] = lo[e]; vb[8 + e] = hi[e]; }
            o[ht] = wmma_f16(pa, vb, o[ht]);
        }
    }

    // ---- epilogue: O / l, store f32 -----------------------------------------
    v8f inv;
    #pragma unroll
    for (int e = 0; e < 8; ++e) inv[e] = 1.0f / lrow[e];
    #pragma unroll
    for (int ht = 0; ht < 16; ++ht)
        #pragma unroll
        for (int e = 0; e < 8; ++e) {
            const int row = q0 + e + 8 * hlf;
            out[((size_t)b * Tc + row) * Hc + ht * 16 + l15] = o[ht][e] * inv[e];
        }
}

// =============================================================================
extern "C" void kernel_launch(void* const* d_in, const int* in_sizes, int n_in,
                              void* d_out, int out_size, void* d_ws, size_t ws_size,
                              hipStream_t stream) {
    const float* E  = (const float*)d_in[0];
    const float* Wq = (const float*)d_in[1];
    const float* Wk = (const float*)d_in[2];
    const float* Wv = (const float*)d_in[3];
    float* out = (float*)d_out;

    // workspace: Qh | Kh | Vt (each B*T*H f16 = 8 MB) | Wt (3 * H*C f16 = 1.5 MB)
    _Float16* Qh  = (_Float16*)d_ws;
    _Float16* Kh  = Qh + (size_t)Bc * Tc * Hc;
    _Float16* Vt  = Kh + (size_t)Bc * Tc * Hc;
    _Float16* Wtq = Vt + (size_t)Bc * Tc * Hc;
    _Float16* Wtk = Wtq + (size_t)Hc * Cc;
    _Float16* Wtv = Wtk + (size_t)Hc * Cc;

    // W -> f16 transposed (bandwidth-trivial)
    {
        const int n = Cc * Hc;                 // 262144 per matrix
        wconv_kernel<<<n / 256, 256, 0, stream>>>(Wq, Wtq);
        wconv_kernel<<<n / 256, 256, 0, stream>>>(Wk, Wtk);
        wconv_kernel<<<n / 256, 256, 0, stream>>>(Wv, Wtv);
    }

    // projections: 1024 row-strips each, 8 waves/block -> 128 blocks per matrix
    {
        const int blocks = ((Bc * Tc) / 16) / 8;
        proj_strip_kernel<0><<<blocks, 256, 0, stream>>>(E, Wtq, Qh);
        proj_strip_kernel<1><<<blocks, 256, 0, stream>>>(E, Wtk, Kh);
        proj_strip_kernel<2><<<blocks, 256, 0, stream>>>(E, Wtv, Vt);
    }

    // attention: 1024 query tiles, 8 waves/block -> 128 blocks
    {
        const int blocks = ((Bc * Tc) / 16) / 8;
        flash_attn_kernel<<<blocks, 256, 0, stream>>>(Qh, Kh, Vt, out);
    }
}
